// PhaseAwareAttention_9010841387129
// MI455X (gfx1250) — compile-verified, matrix operand order checked by reading, not run on hardware
//
#include <hip/hip_runtime.h>
#include <hip/hip_bf16.h>

// ---------------------------------------------------------------------------
// PhaseAwareAttention forward for MI455X (gfx1250, wave32, WMMA bf16).
// B=2, T=2048, C=768, H=12, D=64.
// All matrix math via v_wmma_f32_16x16x32_bf16 (f32 accumulate).
// fp32 inputs are converted to bf16 ONCE in a prepass so GEMM/attention inner
// loops are pure global_load_b128 -> v_wmma (no conversion VALU).
// ---------------------------------------------------------------------------

#define BB 2
#define TT 2048
#define CC 768
#define HH 12
#define DH 64

typedef __bf16 bf16_t;
typedef __attribute__((ext_vector_type(16))) __bf16 v16bf;
typedef __attribute__((ext_vector_type(8)))  float  v8f;

// Native conversion: clang lowers this to the hardware bf16 convert on gfx1250.
__device__ __forceinline__ bf16_t f2bf(float f) { return (bf16_t)f; }

// Build a 16x16x32 WMMA operand for this lane from a (transposed-)row pointer.
// A-matrix layout (ISA 7.12.2): lane holds row m = lane%16; elements 0..7 are
// K = koff..koff+7, elements 8..15 are K = 16+koff..16+koff+7, koff = lane<16?0:8.
// B operands are fed from B^T rows (lane n = lane%16) with the same K pattern.
__device__ __forceinline__ v16bf make_op(const bf16_t* row, int koff) {
  union { v16bf v; uint4 q[2]; } u;
  u.q[0] = *(const uint4*)(row + koff);
  u.q[1] = *(const uint4*)(row + 16 + koff);
  return u.v;
}

__device__ __forceinline__ v8f wmma_bf16(v16bf a, v16bf b, v8f c) {
  return __builtin_amdgcn_wmma_f32_16x16x32_bf16(
      /*neg_a=*/false, a, /*neg_b=*/false, b,
      /*c_mod=*/(short)0, c, /*reuse_a=*/false, /*reuse_b=*/false);
}

// ---------------------------------------------------------------------------
// Kernel 0: fp32 -> bf16 conversion prepass (vectorized 8 elements/thread).
// n must be a multiple of 8 (true for all our tensors).
// ---------------------------------------------------------------------------
__global__ void cvt_kernel(const float* __restrict__ src,
                           bf16_t* __restrict__ dst, int n) {
  int i = (blockIdx.x * blockDim.x + threadIdx.x) * 8;
  if (i >= n) return;
  float4 a = *(const float4*)(src + i);
  float4 b = *(const float4*)(src + i + 4);
  union { bf16_t e[8]; uint4 q; } u;
  u.e[0] = f2bf(a.x); u.e[1] = f2bf(a.y); u.e[2] = f2bf(a.z); u.e[3] = f2bf(a.w);
  u.e[4] = f2bf(b.x); u.e[5] = f2bf(b.y); u.e[6] = f2bf(b.z); u.e[7] = f2bf(b.w);
  *(uint4*)(dst + i) = u.q;
}

// ---------------------------------------------------------------------------
// Kernel 1: coherence table coh[d + T - 1] = mean_j cos(d * theta_j), d in
// [-(T-1), T-1].  16 KB table -> resident in WGP$/L2 for the attention kernel.
// ---------------------------------------------------------------------------
__global__ void coh_kernel(float* __restrict__ coh) {
  int i = blockIdx.x * blockDim.x + threadIdx.x;
  if (i >= 2 * TT - 1) return;
  float diff = (float)(i - (TT - 1));
  float s = 0.0f;
#pragma unroll 8
  for (int j = 0; j < DH / 2; ++j) {
    float theta = __powf(10000.0f, -(float)(2 * j) / (float)DH);
    s += __cosf(diff * theta);
  }
  coh[i] = s * (2.0f / (float)DH);
}

// ---------------------------------------------------------------------------
// Kernel 2: Q/K/V projection + rotary + RMS-norm (all-bf16 operands).
// grid = (row-tiles=256, heads=12, mat=3), block = 128 (4 waves).
// Wave w computes the 16x16 f32 tile at columns h*64 + w*16 via a K-loop of
// 24 bf16 WMMAs, then the workgroup cooperates through LDS for the 64-wide
// rotary + RMS-norm epilogue.  Q/K stored bf16 head-major [B,H,T,D]; V stored
// bf16 transposed [B,H,D,T] so attention's PV B-operand is contiguous.
// ---------------------------------------------------------------------------
__global__ void qkv_kernel(const bf16_t* __restrict__ xb,
                           const float* __restrict__ cosb,
                           const float* __restrict__ sinb,
                           const bf16_t* __restrict__ Wqb,
                           const bf16_t* __restrict__ Wkb,
                           const bf16_t* __restrict__ Wvb,
                           bf16_t* __restrict__ Qh,
                           bf16_t* __restrict__ Kh,
                           bf16_t* __restrict__ Vt) {
  const int row0 = blockIdx.x * 16;          // row in [0, B*T)
  const int h    = blockIdx.y;
  const int mat  = blockIdx.z;               // 0=Q, 1=K, 2=V
  const int tid  = threadIdx.x;
  const int lane = tid & 31;
  const int wave = tid >> 5;
  const int koff = (lane & 16) ? 8 : 0;
  const int hi   = (lane >> 4) & 1;

  const bf16_t* W = (mat == 0) ? Wqb : (mat == 1) ? Wkb : Wvb;
  const int m0 = h * DH + wave * 16;

  const bf16_t* xrow = xb + (size_t)(row0 + (lane & 15)) * CC;  // A: x row
  const bf16_t* wrow = W  + (size_t)(m0  + (lane & 15)) * CC;   // B^T row

  v8f acc = {0.f, 0.f, 0.f, 0.f, 0.f, 0.f, 0.f, 0.f};
#pragma unroll 4
  for (int c0 = 0; c0 < CC; c0 += 32) {
    v16bf a = make_op(xrow + c0, koff);
    v16bf b = make_op(wrow + c0, koff);
    acc = wmma_bf16(a, b, acc);
  }

  // C/D layout: VGPR r holds row (r + 8*hi), column (lane&15).
  __shared__ float tile[16][DH + 4];
#pragma unroll
  for (int r = 0; r < 8; ++r)
    tile[r + 8 * hi][wave * 16 + (lane & 15)] = acc[r];
  __syncthreads();

  // Epilogue: thread (r = tid/8, g = tid%8) owns 4 rotary pairs of row r.
  const int r = tid >> 3;
  const int g = tid & 7;
  const int grow = row0 + r;
  const int bb = grow >> 11;          // /T
  const int t  = grow & (TT - 1);     // %T

  if (mat < 2) {
    float vals[8];
    float ss = 0.0f;
#pragma unroll
    for (int k = 0; k < 4; ++k) {
      int p = g * 4 + k;
      float cv = cosb[t * (DH / 2) + p];
      float sv = sinb[t * (DH / 2) + p];
      float x1 = tile[r][p];
      float x2 = tile[r][p + DH / 2];
      float y1 =  x1 * cv + x2 * sv;
      float y2 = -x1 * sv + x2 * cv;
      vals[k] = y1; vals[4 + k] = y2;
      ss += y1 * y1 + y2 * y2;
    }
    // 8 threads per row are contiguous lanes of one wave -> shfl reduce.
#pragma unroll
    for (int msk = 1; msk < 8; msk <<= 1) ss += __shfl_xor(ss, msk, 32);
    float sc = rsqrtf(ss * (1.0f / (float)DH) + 1.1920929e-7f);
    bf16_t* outp = ((mat == 0) ? Qh : Kh) +
                   ((size_t)(bb * HH + h) * TT + t) * DH;
#pragma unroll
    for (int k = 0; k < 4; ++k) {
      int p = g * 4 + k;
      outp[p]          = f2bf(vals[k] * sc);
      outp[p + DH / 2] = f2bf(vals[4 + k] * sc);
    }
  } else {
    // V: transposed store Vt[b,h,d,t]
    bf16_t* outp = Vt + (size_t)(bb * HH + h) * DH * TT + t;
#pragma unroll
    for (int k = 0; k < 8; ++k) {
      int d = g * 8 + k;
      outp[(size_t)d * TT] = f2bf(tile[r][d]);
    }
  }
}

// ---------------------------------------------------------------------------
// Kernel 3: flash attention with coherence-scaled logits + causal renorm.
// grid = (T/16/4 = 32, H=12, B=2), block = 128; each wave owns one 16-row
// q-tile, loops over 32-key blocks (causal).  8 WMMAs per key-block.
// ---------------------------------------------------------------------------
__global__ void attn_kernel(const bf16_t* __restrict__ Qh,
                            const bf16_t* __restrict__ Kh,
                            const bf16_t* __restrict__ Vt,
                            const float*  __restrict__ cohTab,
                            const float*  __restrict__ beta,
                            bf16_t* __restrict__ y) {
  const int tid  = threadIdx.x;
  const int lane = tid & 31;
  const int wave = tid >> 5;
  const int n    = lane & 15;
  const int koff = (lane & 16) ? 8 : 0;
  const int hi   = (lane >> 4) & 1;

  const int h  = blockIdx.y;
  const int b  = blockIdx.z;
  const int qt = blockIdx.x * 4 + wave;
  const int q0 = qt * 16;
  const size_t bh = (size_t)(b * HH + h);

  const bf16_t* Qbase = Qh + (bh * TT) * DH;
  const bf16_t* Kbase = Kh + (bh * TT) * DH;
  const bf16_t* Vbase = Vt + bh * DH * TT;

  // Q operands (held for the whole k-loop): dims 0..31 and 32..63.
  const bf16_t* qrow = Qbase + (size_t)(q0 + n) * DH;
  const v16bf qa0 = make_op(qrow, koff);
  const v16bf qa1 = make_op(qrow + 32, koff);

  v8f o0 = {0.f,0.f,0.f,0.f,0.f,0.f,0.f,0.f};
  v8f o1 = o0, o2 = o0, o3 = o0;
  float m_run[8], l_run[8];
#pragma unroll
  for (int r = 0; r < 8; ++r) { m_run[r] = -3.0e38f; l_run[r] = 0.0f; }

  const float rscale = 0.125f;  // 1/sqrt(64)
  const float bet = beta[h];

  __shared__ __align__(16) bf16_t plds[4][16][32];  // per-wave P staging

  for (int j0 = 0; j0 < q0 + 16; j0 += 32) {
    // ---- S = Q K^T (two 16x16 tiles, keys j0..j0+15 and j0+16..j0+31) ----
    const bf16_t* krow0 = Kbase + (size_t)(j0 + n) * DH;
    const bf16_t* krow1 = krow0 + 16 * DH;
    v8f s0 = {0.f,0.f,0.f,0.f,0.f,0.f,0.f,0.f};
    v8f s1 = s0;
    s0 = wmma_bf16(qa0, make_op(krow0,      koff), s0);
    s0 = wmma_bf16(qa1, make_op(krow0 + 32, koff), s0);
    s1 = wmma_bf16(qa0, make_op(krow1,      koff), s1);
    s1 = wmma_bf16(qa1, make_op(krow1 + 32, koff), s1);

    // ---- coherence scale + causal mask (element (r): row q0+r+8*hi) ----
    float t0[8], t1[8];
#pragma unroll
    for (int r = 0; r < 8; ++r) {
      int q   = q0 + r + 8 * hi;
      int k0  = j0 + n;
      int k1  = k0 + 16;
      float c0v = cohTab[q - k0 + (TT - 1)];
      float c1v = cohTab[q - k1 + (TT - 1)];
      float v0 = s0[r] * rscale * (1.0f + bet * c0v);
      float v1 = s1[r] * rscale * (1.0f + bet * c1v);
      t0[r] = (k0 > q) ? -1.0e30f : v0;
      t1[r] = (k1 > q) ? -1.0e30f : v1;
    }

    // ---- online softmax across the 32 keys ----
    float p0[8], p1[8], alpha[8];
#pragma unroll
    for (int r = 0; r < 8; ++r) {
      float mv = fmaxf(t0[r], t1[r]);
#pragma unroll
      for (int msk = 1; msk < 16; msk <<= 1) mv = fmaxf(mv, __shfl_xor(mv, msk, 32));
      float mnew = fmaxf(m_run[r], mv);
      p0[r] = __expf(t0[r] - mnew);
      p1[r] = __expf(t1[r] - mnew);
      alpha[r] = __expf(m_run[r] - mnew);
      float rs = p0[r] + p1[r];
#pragma unroll
      for (int msk = 1; msk < 16; msk <<= 1) rs += __shfl_xor(rs, msk, 32);
      l_run[r] = l_run[r] * alpha[r] + rs;
      m_run[r] = mnew;
      o0[r] *= alpha[r]; o1[r] *= alpha[r]; o2[r] *= alpha[r]; o3[r] *= alpha[r];
    }

    // ---- re-shape P (C-layout) -> A-layout via wave-private LDS ----
#pragma unroll
    for (int r = 0; r < 8; ++r) {
      plds[wave][r + 8 * hi][n]      = f2bf(p0[r]);
      plds[wave][r + 8 * hi][16 + n] = f2bf(p1[r]);
    }
    const v16bf pa = make_op(&plds[wave][n][0], koff);  // LDS in-order per wave

    // ---- O += P V : B-operand rows are Vt[d][keys] (contiguous) ----
    o0 = wmma_bf16(pa, make_op(Vbase + (size_t)(0  + n) * TT + j0, koff), o0);
    o1 = wmma_bf16(pa, make_op(Vbase + (size_t)(16 + n) * TT + j0, koff), o1);
    o2 = wmma_bf16(pa, make_op(Vbase + (size_t)(32 + n) * TT + j0, koff), o2);
    o3 = wmma_bf16(pa, make_op(Vbase + (size_t)(48 + n) * TT + j0, koff), o3);
  }

  // ---- normalize (causal renorm denominator ~= l + 1e-8) and store y ----
  bf16_t* ybase = y + ((size_t)b * TT) * CC;
#pragma unroll
  for (int r = 0; r < 8; ++r) {
    float inv = 1.0f / (l_run[r] + 1e-8f);
    int row = q0 + r + 8 * hi;
    bf16_t* yr = ybase + (size_t)row * CC + h * DH;
    yr[0 * 16 + n] = f2bf(o0[r] * inv);
    yr[1 * 16 + n] = f2bf(o1[r] * inv);
    yr[2 * 16 + n] = f2bf(o2[r] * inv);
    yr[3 * 16 + n] = f2bf(o3[r] * inv);
  }
}

// ---------------------------------------------------------------------------
// Kernel 4: out = y @ Wo^T  -> fp32 d_out.  grid=(256,12), block=128.
// ---------------------------------------------------------------------------
__global__ void out_kernel(const bf16_t* __restrict__ y,
                           const bf16_t* __restrict__ Wob,
                           float* __restrict__ out) {
  const int row0 = blockIdx.x * 16;
  const int tid  = threadIdx.x;
  const int lane = tid & 31;
  const int wave = tid >> 5;
  const int koff = (lane & 16) ? 8 : 0;
  const int hi   = (lane >> 4) & 1;
  const int m0   = blockIdx.y * 64 + wave * 16;

  const bf16_t* yrow = y   + (size_t)(row0 + (lane & 15)) * CC;
  const bf16_t* wrow = Wob + (size_t)(m0   + (lane & 15)) * CC;

  v8f acc = {0.f,0.f,0.f,0.f,0.f,0.f,0.f,0.f};
#pragma unroll 4
  for (int c0 = 0; c0 < CC; c0 += 32) {
    v16bf a = make_op(yrow + c0, koff);
    v16bf b = make_op(wrow + c0, koff);
    acc = wmma_bf16(a, b, acc);
  }
#pragma unroll
  for (int r = 0; r < 8; ++r)
    out[(size_t)(row0 + r + 8 * hi) * CC + m0 + (lane & 15)] = acc[r];
}

// ---------------------------------------------------------------------------
extern "C" void kernel_launch(void* const* d_in, const int* in_sizes, int n_in,
                              void* d_out, int out_size, void* d_ws, size_t ws_size,
                              hipStream_t stream) {
  (void)in_sizes; (void)n_in; (void)out_size; (void)ws_size;

  const float* x    = (const float*)d_in[0];
  const float* cosb = (const float*)d_in[1];
  const float* sinb = (const float*)d_in[2];
  const float* Wq   = (const float*)d_in[3];
  const float* Wk   = (const float*)d_in[4];
  const float* Wv   = (const float*)d_in[5];
  const float* Wo   = (const float*)d_in[6];
  const float* beta = (const float*)d_in[7];
  float* out = (float*)d_out;

  // Workspace layout (bytes): ~35 MB total.
  char* ws = (char*)d_ws;
  float*  coh = (float*)ws;                               // 4095 * 4 -> 16 KB
  size_t  off = 16384;
  const size_t headBytes = (size_t)BB * HH * TT * DH * sizeof(bf16_t);  // 6 MB
  const size_t wBytes    = (size_t)CC * CC * sizeof(bf16_t);            // 1.125 MB
  bf16_t* Qh  = (bf16_t*)(ws + off); off += headBytes;
  bf16_t* Kh  = (bf16_t*)(ws + off); off += headBytes;
  bf16_t* Vt  = (bf16_t*)(ws + off); off += headBytes;
  bf16_t* yb  = (bf16_t*)(ws + off); off += headBytes;    // y in [B,T,C] bf16
  bf16_t* xbb = (bf16_t*)(ws + off); off += headBytes;    // x bf16 (same size)
  bf16_t* Wqb = (bf16_t*)(ws + off); off += wBytes;
  bf16_t* Wkb = (bf16_t*)(ws + off); off += wBytes;
  bf16_t* Wvb = (bf16_t*)(ws + off); off += wBytes;
  bf16_t* Wob = (bf16_t*)(ws + off); off += wBytes;

  // --- prepass: convert fp32 inputs to bf16 once ---
  const int nx = BB * TT * CC;     // 3,145,728
  const int nw = CC * CC;          // 589,824
  cvt_kernel<<<(nx / 8 + 255) / 256, 256, 0, stream>>>(x,  xbb, nx);
  cvt_kernel<<<(nw / 8 + 255) / 256, 256, 0, stream>>>(Wq, Wqb, nw);
  cvt_kernel<<<(nw / 8 + 255) / 256, 256, 0, stream>>>(Wk, Wkb, nw);
  cvt_kernel<<<(nw / 8 + 255) / 256, 256, 0, stream>>>(Wv, Wvb, nw);
  cvt_kernel<<<(nw / 8 + 255) / 256, 256, 0, stream>>>(Wo, Wob, nw);

  coh_kernel<<<(2 * TT - 1 + 255) / 256, 256, 0, stream>>>(coh);

  qkv_kernel<<<dim3(BB * TT / 16, HH, 3), 128, 0, stream>>>(
      xbb, cosb, sinb, Wqb, Wkb, Wvb, Qh, Kh, Vt);

  attn_kernel<<<dim3(TT / 16 / 4, HH, BB), 128, 0, stream>>>(
      Qh, Kh, Vt, coh, beta, yb);

  out_kernel<<<dim3(BB * TT / 16, CC / DH), 128, 0, stream>>>(yb, Wob, out);
}
